// Trans_25512105738365
// MI455X (gfx1250) — matrix operand order repaired and checked by reading, not compiled, not run on hardware
//
#include <hip/hip_runtime.h>

// ---------------- model dims ----------------
#define BATCH  1024
#define SEQ    50
#define NTOK   (BATCH*SEQ)     // 51200
#define DMODEL 200
#define DPAD   256             // padded model dim (zero-padded cols/rows)
#define NHEAD  8
#define DHEAD  25
#define FFDIM  512
#define NLAYER 8
#define CONVN  6912            // 3*48*48, multiple of 32

typedef unsigned short bf16_t;
typedef __attribute__((ext_vector_type(8)))  unsigned short u16x8;
typedef __attribute__((ext_vector_type(8)))  __bf16 v8bf;
typedef __attribute__((ext_vector_type(16))) __bf16 v16bf;
typedef __attribute__((ext_vector_type(8)))  float  v8f;
typedef __attribute__((ext_vector_type(4)))  unsigned int u32x4;
typedef __attribute__((ext_vector_type(8)))  int i32x8;
typedef __attribute__((ext_vector_type(4)))  int i32x4;

__device__ __forceinline__ bf16_t f2bf(float f) {
  union { float f; unsigned u; } x; x.f = f;
  unsigned r = x.u + 0x7FFFu + ((x.u >> 16) & 1u);   // round-to-nearest-even
  return (bf16_t)(r >> 16);
}
__device__ __forceinline__ float bf2f(bf16_t h) {
  union { unsigned u; float f; } x; x.u = ((unsigned)h) << 16;
  return x.f;
}

// =====================================================================
// TDM 2D tile load: global (row-major, strided) -> LDS, with LDS row
// padding (16 DWORDs data + 4 DWORDs pad => 40-half LDS row stride).
// Descriptor packed per CDNA5 ISA D# (group0 128b, group1 256b).
// Issue from ONE wave only; completion via s_wait_tensorcnt.
// This toolchain exposes the 6-arg builtin:
//   (u32x4 g0, i32x8 g1, i32x4 g2, i32x4 g3, i32x8 extra, i32 cpol)
// Groups 2/3/extra are unused for a 2D descriptor -> zero-filled.
// =====================================================================
__device__ __forceinline__ void tdm_load_2d_pad(const void* gaddr, void* lds,
                                                unsigned rowHalves,   // tile_dim0 (2B units)
                                                unsigned rows,        // tile_dim1
                                                unsigned strideElems) // tensor_dim0_stride
{
  unsigned long long ga = (unsigned long long)(uintptr_t)gaddr;
  unsigned lo = (unsigned)(uintptr_t)lds;   // low 32 bits of generic LDS ptr = LDS byte offset
  u32x4 g0;
  g0[0] = 1u;                                              // count=1, user descriptor
  g0[1] = lo;                                              // lds_addr
  g0[2] = (unsigned)(ga & 0xFFFFFFFFull);                  // global_addr[31:0]
  g0[3] = (unsigned)((ga >> 32) & 0x1FFFFFFull) | (2u << 30); // global_addr[56:32], type=2
  i32x8 g1;
  // data_size=1 (2B), pad_enable, pad_interval=3 (16 DWORDs), pad_amount=3 (4 DWORDs)
  g1[0] = (int)((1u << 16) | (1u << 20) | (3u << 22) | (3u << 25));
  g1[1] = (int)((rowHalves & 0xFFFFu) << 16);              // tensor_dim0[15:0] @ bits63:48
  g1[2] = (int)(((rowHalves >> 16) & 0xFFFFu) | ((rows & 0xFFFFu) << 16)); // dim0 hi | dim1 lo
  g1[3] = (int)(((rows >> 16) & 0xFFFFu) | ((rowHalves & 0xFFFFu) << 16)); // dim1 hi | tile_dim0
  g1[4] = (int)(rows & 0xFFFFu);                           // tile_dim1 | tile_dim2=0
  g1[5] = (int)strideElems;                                // tensor_dim0_stride[31:0]
  g1[6] = 0;                                               // stride hi | dim1_stride lo (unused, 2D)
  g1[7] = 0;
  i32x4 z4 = {0, 0, 0, 0};
  i32x8 z8 = {0, 0, 0, 0, 0, 0, 0, 0};
  __builtin_amdgcn_tensor_load_to_lds(g0, g1, z4, z4, z8, 0);
}

// =====================================================================
// Tiled bf16 WMMA GEMM:  C[M,N] = A[M,K](bf16) * B[K,N](bf16) + bias
// Block: 128 threads (4 waves), 128x64 C tile, K staged in 32-chunks.
// Wave (2x2 grid) computes a 64x32 subtile as 4x2 v_wmma_f32_16x16x32_bf16
// (8 WMMA per K-step from 6 fragment loads).
// A tile staged by the Tensor Data Mover (async, padded LDS rows);
// B tile staged transposed by vector loads.
// Requires M%128==0, N%64==0, K%32==0, 16B-aligned rows.
// OB: 1 -> bf16 output, 0 -> f32 output.  RL: ReLU.
// =====================================================================
#define GBM 128
#define GBN 64
#define GBK 32
#define LDSS 40   // LDS row stride in halves (80B: 16B aligned, bank-spread)

template<int OB, int RL>
__global__ __launch_bounds__(128)
void gemm_bf16(const bf16_t* __restrict__ A, int lda,
               const bf16_t* __restrict__ B, int ldb,
               void* __restrict__ C, int ldc,
               const float* __restrict__ bias, int bias_n,
               int M, int N, int K)
{
  __shared__ bf16_t As[GBM * LDSS];   // TDM-written, 40-half row stride
  __shared__ bf16_t Bs[GBN * LDSS];   // transposed: Bs[n][k]

  const int tid  = threadIdx.x;
  const int wave = tid >> 5;
  const int lane = tid & 31;
  const int bm = blockIdx.y * GBM;
  const int bn = blockIdx.x * GBN;
  const int wm = (wave >> 1) * 64;    // wave row offset (2 waves in M)
  const int wn = (wave & 1) * 32;     // wave col offset (2 waves in N)
  const int lm = lane & 15;           // fragment row/col within 16
  const int lg = lane >> 4;           // lane group 0/1

  const v8f zero8 = {0.f,0.f,0.f,0.f,0.f,0.f,0.f,0.f};
  v8f acc[4][2];
#pragma unroll
  for (int i = 0; i < 4; ++i) { acc[i][0] = zero8; acc[i][1] = zero8; }

  for (int k0 = 0; k0 < K; k0 += GBK) {
    // ---- stage B tile transposed: Bs[n][k] = B[k0+k][bn+n] ----
    for (int idx = tid; idx < GBK * GBN; idx += 128) {
      int k = idx >> 6, n = idx & 63;
      Bs[n * LDSS + k] = B[(size_t)(k0 + k) * ldb + bn + n];
    }
    // prefetch next B tile rows (gfx1250 global_prefetch_b8)
    if (k0 + GBK < K)
      __builtin_prefetch(B + (size_t)(k0 + GBK + (tid >> 2)) * ldb + bn, 0, 1);
    // ---- stage A tile (128 rows x 32 halves) via Tensor Data Mover ----
    if (wave == 0) {
      tdm_load_2d_pad(A + (size_t)bm * lda + k0, As, GBK, GBM, (unsigned)lda);
      __builtin_amdgcn_s_wait_tensorcnt(0);   // tile complete before barrier
    }
    __syncthreads();

    // ---- build fragments per ISA VGPR layouts ----
    union FRG { v16bf v; v8bf h[2]; };
    FRG af[4], bfr[2];
#pragma unroll
    for (int i = 0; i < 4; ++i) {
      // A 16x32 bf16: lane m=lm, kbase=lg*8; v0-3:K=kb..kb+7, v4-7:K=16+kb..
      const bf16_t* rp = As + (wm + i * 16 + lm) * LDSS;
      af[i].h[0] = *(const v8bf*)(rp + lg * 8);
      af[i].h[1] = *(const v8bf*)(rp + 16 + lg * 8);
    }
#pragma unroll
    for (int j = 0; j < 2; ++j) {
      // B 32x16 bf16: lane n=lm, kbase=lg*8; v0-3:K=kb..kb+7, v4-7:K=16+kb..
      const bf16_t* cp = Bs + (wn + j * 16 + lm) * LDSS + lg * 8;
      bfr[j].h[0] = *(const v8bf*)(cp);
      bfr[j].h[1] = *(const v8bf*)(cp + 16);
    }
#pragma unroll
    for (int i = 0; i < 4; ++i)
#pragma unroll
      for (int j = 0; j < 2; ++j)
        acc[i][j] = __builtin_amdgcn_wmma_f32_16x16x32_bf16(
            false, af[i].v, false, bfr[j].v, (short)0, acc[i][j], false, false);
    __syncthreads();
  }

  // ---- epilogue: C layout lane holds row = lg*8 + r, col = lm ----
#pragma unroll
  for (int i = 0; i < 4; ++i) {
#pragma unroll
    for (int j = 0; j < 2; ++j) {
      const int col = bn + wn + j * 16 + lm;
      const float bv = bias ? ((col < bias_n) ? bias[col] : 0.f) : 0.f;
#pragma unroll
      for (int r = 0; r < 8; ++r) {
        const int row = bm + wm + i * 16 + lg * 8 + r;
        float v = acc[i][j][r] + bv;
        if (RL) v = fmaxf(v, 0.f);
        const size_t ix = (size_t)row * ldc + col;
        if (OB) ((bf16_t*)C)[ix] = f2bf(v);
        else    ((float*)C)[ix]  = v;
      }
    }
  }
}

// =====================================================================
// f32 [K,N] -> bf16 [Kp,Np] zero-padded weight conversion
// =====================================================================
__global__ void convert_pad(const float* __restrict__ src, bf16_t* __restrict__ dst,
                            int K, int N, int Kp, int Np)
{
  int idx = blockIdx.x * blockDim.x + threadIdx.x;
  if (idx >= Kp * Np) return;
  int k = idx / Np, n = idx % Np;
  float v = (k < K && n < N) ? src[k * N + n] : 0.f;
  dst[idx] = f2bf(v);
}

// =====================================================================
// Embedding gather + positional add + LayerNorm -> bf16 padded [NTOK,256]
// =====================================================================
__global__ __launch_bounds__(256)
void embed_ln(const int* __restrict__ idx, const float* __restrict__ W,
              const float* __restrict__ P, const float* __restrict__ g,
              const float* __restrict__ b, bf16_t* __restrict__ X)
{
  int tok = blockIdx.x, t = threadIdx.x;
  int s = tok % SEQ;
  int id = idx[tok];
  float v = 0.f;
  if (t < DMODEL) v = W[(size_t)id * DMODEL + t] + P[s * DMODEL + t];
  __shared__ float red[256];
  red[t] = (t < DMODEL) ? v : 0.f;
  __syncthreads();
  for (int o = 128; o > 0; o >>= 1) { if (t < o) red[t] += red[t + o]; __syncthreads(); }
  float mean = red[0] * (1.f / DMODEL);
  __syncthreads();
  float d = (t < DMODEL) ? (v - mean) : 0.f;
  red[t] = d * d;
  __syncthreads();
  for (int o = 128; o > 0; o >>= 1) { if (t < o) red[t] += red[t + o]; __syncthreads(); }
  float rs = rsqrtf(red[0] * (1.f / DMODEL) + 1e-12f);
  X[(size_t)tok * DPAD + t] = (t < DMODEL) ? f2bf(g[t] * d * rs + b[t]) : (bf16_t)0;
}

// =====================================================================
// out = LN(res(bf16) + y(f32)) -> bf16 padded (pads zeroed)
// =====================================================================
__global__ __launch_bounds__(256)
void residual_ln(const bf16_t* __restrict__ res, const float* __restrict__ y,
                 const float* __restrict__ g, const float* __restrict__ b,
                 bf16_t* __restrict__ out)
{
  int tok = blockIdx.x, t = threadIdx.x;
  size_t base = (size_t)tok * DPAD;
  float v = 0.f;
  if (t < DMODEL) v = bf2f(res[base + t]) + y[base + t];
  __shared__ float red[256];
  red[t] = (t < DMODEL) ? v : 0.f;
  __syncthreads();
  for (int o = 128; o > 0; o >>= 1) { if (t < o) red[t] += red[t + o]; __syncthreads(); }
  float mean = red[0] * (1.f / DMODEL);
  __syncthreads();
  float d = (t < DMODEL) ? (v - mean) : 0.f;
  red[t] = d * d;
  __syncthreads();
  for (int o = 128; o > 0; o >>= 1) { if (t < o) red[t] += red[t + o]; __syncthreads(); }
  float rs = rsqrtf(red[0] * (1.f / DMODEL) + 1e-12f);
  out[base + t] = (t < DMODEL) ? f2bf(g[t] * d * rs + b[t]) : (bf16_t)0;
}

// =====================================================================
// Per-(b,h) masked softmax attention.  DH=25, S=50: VALU f32 in LDS.
// Writes ctx (bf16) into padded activation layout at head slice.
// =====================================================================
__global__ __launch_bounds__(128)
void attention_kernel(const bf16_t* __restrict__ Q, const bf16_t* __restrict__ K,
                      const bf16_t* __restrict__ V, const int* __restrict__ mask,
                      bf16_t* __restrict__ CTX)
{
  __shared__ float qs[SEQ * DHEAD], ks[SEQ * DHEAD], vs[SEQ * DHEAD];
  __shared__ float sc[SEQ * SEQ];
  __shared__ float mk[64];
  int bh = blockIdx.x;
  int b = bh >> 3, h = bh & 7;
  int t = threadIdx.x;
  size_t base = (size_t)b * SEQ * DPAD + h * DHEAD;

  for (int i = t; i < SEQ * DHEAD; i += 128) {
    int s = i / DHEAD, d = i % DHEAD;
    size_t a = base + (size_t)s * DPAD + d;
    qs[i] = bf2f(Q[a]); ks[i] = bf2f(K[a]); vs[i] = bf2f(V[a]);
  }
  if (t < SEQ) mk[t] = (1.f - (float)mask[b * SEQ + t]) * -10000.f;
  __syncthreads();

  for (int i = t; i < SEQ * SEQ; i += 128) {
    int q = i / SEQ, k = i % SEQ;
    float s = 0.f;
    for (int d = 0; d < DHEAD; ++d) s += qs[q * DHEAD + d] * ks[k * DHEAD + d];
    sc[i] = s * 0.2f + mk[k];              // 1/sqrt(25) = 0.2
  }
  __syncthreads();

  if (t < SEQ) {
    float m = -1e30f;
    for (int j = 0; j < SEQ; ++j) m = fmaxf(m, sc[t * SEQ + j]);
    float su = 0.f;
    for (int j = 0; j < SEQ; ++j) { float e = __expf(sc[t * SEQ + j] - m); sc[t * SEQ + j] = e; su += e; }
    float inv = 1.f / su;
    for (int j = 0; j < SEQ; ++j) sc[t * SEQ + j] *= inv;
  }
  __syncthreads();

  for (int i = t; i < SEQ * DHEAD; i += 128) {
    int q = i / DHEAD, d = i % DHEAD;
    float s = 0.f;
    for (int k = 0; k < SEQ; ++k) s += sc[q * SEQ + k] * vs[k * DHEAD + d];
    CTX[base + (size_t)q * DPAD + d] = f2bf(s);
  }
}

// zero columns 200..255 of a [NTOK,256] bf16 buffer
__global__ void zero_pad_cols(bf16_t* __restrict__ X)
{
  int tok = blockIdx.x, t = threadIdx.x;
  if (t < DPAD - DMODEL) X[(size_t)tok * DPAD + DMODEL + t] = 0;
}

// =====================================================================
// iv[b,i,j] = dot(xd[b,i,:], xe[b,j,:]) over D=200
// =====================================================================
__global__ __launch_bounds__(256)
void interaction_kernel(const bf16_t* __restrict__ Xd, const bf16_t* __restrict__ Xe,
                        float* __restrict__ iv)
{
  __shared__ bf16_t sd[SEQ * DMODEL], se[SEQ * DMODEL];
  int b = blockIdx.x, t = threadIdx.x;
  size_t base = (size_t)b * SEQ * DPAD;
  for (int i = t; i < SEQ * DMODEL; i += 256) {
    int s = i / DMODEL, d = i % DMODEL;
    sd[i] = Xd[base + (size_t)s * DPAD + d];
    se[i] = Xe[base + (size_t)s * DPAD + d];
  }
  __syncthreads();
  for (int i = t; i < SEQ * SEQ; i += 256) {
    int q = i / SEQ, j = i % SEQ;
    float acc = 0.f;
    for (int d = 0; d < DMODEL; ++d)
      acc += bf2f(sd[q * DMODEL + d]) * bf2f(se[j * DMODEL + d]);
    iv[(size_t)b * SEQ * SEQ + i] = acc;
  }
}

// =====================================================================
// 3x3 VALID conv, 3 out channels, + bias, flatten NCHW -> bf16 [B,6912]
// =====================================================================
__global__ __launch_bounds__(256)
void conv3_kernel(const float* __restrict__ iv, const float* __restrict__ cw,
                  const float* __restrict__ cb, bf16_t* __restrict__ out)
{
  int b = blockIdx.x;
  for (int i = threadIdx.x; i < CONVN; i += 256) {
    int c = i / 2304, r = i % 2304, y = r / 48, x = r % 48;
    float s = cb[c];
    for (int dy = 0; dy < 3; ++dy)
      for (int dx = 0; dx < 3; ++dx)
        s += cw[c * 9 + dy * 3 + dx] * iv[(size_t)b * 2500 + (y + dy) * 50 + (x + dx)];
    out[(size_t)b * CONVN + i] = f2bf(s);
  }
}

// eval-mode BatchNorm1d over [BATCH, nfeat]
template<int OB>
__global__ void bn_kernel(const float* __restrict__ x, const float* __restrict__ g,
                          const float* __restrict__ b, const float* __restrict__ m,
                          const float* __restrict__ v, void* __restrict__ out,
                          int nfeat, int total)
{
  int i = blockIdx.x * blockDim.x + threadIdx.x;
  if (i >= total) return;
  int c = i % nfeat;
  float y = g[c] * (x[i] - m[c]) * rsqrtf(v[c] + 1e-5f) + b[c];
  if (OB) ((bf16_t*)out)[i] = f2bf(y);
  else    ((float*)out)[i]  = y;
}

// h3 = relu(h2 @ W3 + b3) (64->32);  out = h3 @ W4 + b4  (per sample)
__global__ __launch_bounds__(64)
void head_tail(const float* __restrict__ h2, const float* __restrict__ W3,
               const float* __restrict__ b3, const float* __restrict__ W4,
               const float* __restrict__ b4, float* __restrict__ out)
{
  __shared__ float h[64], t3[32];
  int b = blockIdx.x, t = threadIdx.x;
  h[t] = h2[b * 64 + t];
  __syncthreads();
  if (t < 32) {
    float s = b3[t];
    for (int i = 0; i < 64; ++i) s += h[i] * W3[i * 32 + t];
    t3[t] = fmaxf(s, 0.f);
  }
  __syncthreads();
  if (t == 0) {
    float s = b4[0];
    for (int i = 0; i < 32; ++i) s += t3[i] * W4[i];
    out[b] = s;
  }
}

// =====================================================================
// Host orchestration
// =====================================================================
// Input index map (jax pytree flatten: dicts sorted by key, ASCII order):
//  0 Drug  1 DrugMask  2 SE  3 SEMask
//  4 W1 [6912,512]  5 W2 [512,64]  6 W3 [64,32]  7 W4 [32,1]
//  8 b1[512]  9 b2[64]  10 b3[32]  11 b4[1]
// 12 bn1_b 13 bn1_g 14 bn1_m 15 bn1_v   16 bn2_b 17 bn2_g 18 bn2_m 19 bn2_v
// 20 conv_b[3] 21 conv_w[3,1,3,3]
// 22 emb_d.P 23 emb_d.W 24 emb_d.b 25 emb_d.g
// 26 emb_e.P 27 emb_e.W 28 emb_e.b 29 emb_e.g
// 30.. enc_d: Wi,Wk,Wo,Wp,Wq,Wv,b1,b2,bi,bk,bo,bp,bq,bv,g1,g2  (each [L,...])
// 46.. enc_e: same order

struct EncParams {
  const bf16_t* wenc;   // this encoder's 8-layer padded bf16 weight block
  const float *b1, *b2, *bi, *bk, *bo, *bp, *bq, *bv, *g1, *g2;
};

// per-layer weight block layout (in halves)
#define WQ_OFF 0
#define WK_OFF 65536
#define WV_OFF 131072
#define WO_OFF 196608
#define WI_OFF 262144
#define WP_OFF 393216
#define LAYER_HALVES 524288

static void run_encoder(bf16_t* X, const int* mask, const EncParams& P,
                        bf16_t* Qb, bf16_t* Kb, bf16_t* Vb, bf16_t* Ab,
                        float* Yf, bf16_t* Ib, hipStream_t s)
{
  const dim3 g256(DPAD / GBN, NTOK / GBM);    // (4, 400)
  const dim3 g512(FFDIM / GBN, NTOK / GBM);   // (8, 400)
  for (int l = 0; l < NLAYER; ++l) {
    const bf16_t* wl = P.wenc + (size_t)l * LAYER_HALVES;
    gemm_bf16<1,0><<<g256,128,0,s>>>(X, DPAD, wl + WQ_OFF, DPAD, Qb, DPAD,
                                     P.bq + l * DMODEL, DMODEL, NTOK, DPAD, DPAD);
    gemm_bf16<1,0><<<g256,128,0,s>>>(X, DPAD, wl + WK_OFF, DPAD, Kb, DPAD,
                                     P.bk + l * DMODEL, DMODEL, NTOK, DPAD, DPAD);
    gemm_bf16<1,0><<<g256,128,0,s>>>(X, DPAD, wl + WV_OFF, DPAD, Vb, DPAD,
                                     P.bv + l * DMODEL, DMODEL, NTOK, DPAD, DPAD);
    attention_kernel<<<BATCH * NHEAD, 128, 0, s>>>(Qb, Kb, Vb, mask, Ab);
    gemm_bf16<0,0><<<g256,128,0,s>>>(Ab, DPAD, wl + WO_OFF, DPAD, Yf, DPAD,
                                     P.bo + l * DMODEL, DMODEL, NTOK, DPAD, DPAD);
    residual_ln<<<NTOK,256,0,s>>>(X, Yf, P.g1 + l * DMODEL, P.b1 + l * DMODEL, Ab);
    gemm_bf16<1,1><<<g512,128,0,s>>>(Ab, DPAD, wl + WI_OFF, FFDIM, Ib, FFDIM,
                                     P.bi + l * FFDIM, FFDIM, NTOK, FFDIM, DPAD);
    gemm_bf16<0,0><<<g256,128,0,s>>>(Ib, FFDIM, wl + WP_OFF, DPAD, Yf, DPAD,
                                     P.bp + l * DMODEL, DMODEL, NTOK, DPAD, FFDIM);
    residual_ln<<<NTOK,256,0,s>>>(Ab, Yf, P.g2 + l * DMODEL, P.b2 + l * DMODEL, X);
  }
}

extern "C" void kernel_launch(void* const* d_in, const int* in_sizes, int n_in,
                              void* d_out, int out_size, void* d_ws, size_t ws_size,
                              hipStream_t stream)
{
  (void)in_sizes; (void)out_size;
  if (n_in < 62) return;

  const int*   Drug     = (const int*)d_in[0];
  const int*   DrugMask = (const int*)d_in[1];
  const int*   SE       = (const int*)d_in[2];
  const int*   SEMask   = (const int*)d_in[3];
  const float* W1 = (const float*)d_in[4];
  const float* W2 = (const float*)d_in[5];
  const float* W3 = (const float*)d_in[6];
  const float* W4 = (const float*)d_in[7];
  const float* b1f = (const float*)d_in[8];
  const float* b2f = (const float*)d_in[9];
  const float* b3f = (const float*)d_in[10];
  const float* b4f = (const float*)d_in[11];
  const float* bn1_b = (const float*)d_in[12];
  const float* bn1_g = (const float*)d_in[13];
  const float* bn1_m = (const float*)d_in[14];
  const float* bn1_v = (const float*)d_in[15];
  const float* bn2_b = (const float*)d_in[16];
  const float* bn2_g = (const float*)d_in[17];
  const float* bn2_m = (const float*)d_in[18];
  const float* bn2_v = (const float*)d_in[19];
  const float* conv_b = (const float*)d_in[20];
  const float* conv_w = (const float*)d_in[21];

  // ---------- workspace layout ----------
  char* ws = (char*)d_ws;
  size_t off = 0;
  auto carve = [&](size_t bytes) -> char* {
    char* p = ws + off;
    off = (off + bytes + 255) & ~(size_t)255;
    return p;
  };
  const size_t ACT_B = (size_t)NTOK * DPAD * sizeof(bf16_t);   // 26,214,400
  bf16_t* Xd   = (bf16_t*)carve(ACT_B);
  bf16_t* Xe   = (bf16_t*)carve(ACT_B);
  bf16_t* Ab   = (bf16_t*)carve(ACT_B);
  bf16_t* Qb   = (bf16_t*)carve(2 * ACT_B);          // Q + K contiguous
  bf16_t* Kb   = Qb + (size_t)NTOK * DPAD;
  bf16_t* Ib   = Qb;                                 // INTER aliases Q+K (dead then)
  bf16_t* Vb   = (bf16_t*)carve(ACT_B);
  float*  Yf   = (float*)carve((size_t)NTOK * DPAD * sizeof(float));
  bf16_t* WENC = (bf16_t*)carve((size_t)2 * NLAYER * LAYER_HALVES * sizeof(bf16_t));
  bf16_t* W1bf = (bf16_t*)carve((size_t)CONVN * 512 * sizeof(bf16_t));
  bf16_t* W2bf = (bf16_t*)carve((size_t)512 * 64 * sizeof(bf16_t));
  float*  iv   = (float*)carve((size_t)BATCH * SEQ * SEQ * sizeof(float));
  bf16_t* cvo  = (bf16_t*)carve((size_t)BATCH * CONVN * sizeof(bf16_t));
  float*  h1   = (float*)carve((size_t)BATCH * 512 * sizeof(float));
  bf16_t* h1bn = (bf16_t*)carve((size_t)BATCH * 512 * sizeof(bf16_t));
  float*  h2   = (float*)carve((size_t)BATCH * 64 * sizeof(float));
  float*  h2bn = (float*)carve((size_t)BATCH * 64 * sizeof(float));
  if (off > ws_size) return;   // workspace too small: bail safely

  // ---------- weight prep: f32 -> bf16, zero-padded to WMMA tiles ----------
  for (int e = 0; e < 2; ++e) {
    int base = 30 + 16 * e;
    const float* Wi = (const float*)d_in[base + 0];
    const float* Wk = (const float*)d_in[base + 1];
    const float* Wo = (const float*)d_in[base + 2];
    const float* Wp = (const float*)d_in[base + 3];
    const float* Wq = (const float*)d_in[base + 4];
    const float* Wv = (const float*)d_in[base + 5];
    bf16_t* enc = WENC + (size_t)e * NLAYER * LAYER_HALVES;
    for (int l = 0; l < NLAYER; ++l) {
      bf16_t* wl = enc + (size_t)l * LAYER_HALVES;
      convert_pad<<<256,256,0,stream>>>(Wq + (size_t)l*DMODEL*DMODEL, wl + WQ_OFF, DMODEL, DMODEL, DPAD, DPAD);
      convert_pad<<<256,256,0,stream>>>(Wk + (size_t)l*DMODEL*DMODEL, wl + WK_OFF, DMODEL, DMODEL, DPAD, DPAD);
      convert_pad<<<256,256,0,stream>>>(Wv + (size_t)l*DMODEL*DMODEL, wl + WV_OFF, DMODEL, DMODEL, DPAD, DPAD);
      convert_pad<<<256,256,0,stream>>>(Wo + (size_t)l*DMODEL*DMODEL, wl + WO_OFF, DMODEL, DMODEL, DPAD, DPAD);
      convert_pad<<<512,256,0,stream>>>(Wi + (size_t)l*DMODEL*FFDIM,  wl + WI_OFF, DMODEL, FFDIM, DPAD, FFDIM);
      convert_pad<<<512,256,0,stream>>>(Wp + (size_t)l*FFDIM*DMODEL,  wl + WP_OFF, FFDIM, DMODEL, FFDIM, DPAD);
    }
  }
  convert_pad<<<(CONVN*512)/256,256,0,stream>>>(W1, W1bf, CONVN, 512, CONVN, 512);
  convert_pad<<<(512*64)/256,256,0,stream>>>(W2, W2bf, 512, 64, 512, 64);

  // ---------- embeddings + LN ----------
  embed_ln<<<NTOK,256,0,stream>>>(Drug, (const float*)d_in[23], (const float*)d_in[22],
                                  (const float*)d_in[25], (const float*)d_in[24], Xd);
  embed_ln<<<NTOK,256,0,stream>>>(SE,   (const float*)d_in[27], (const float*)d_in[26],
                                  (const float*)d_in[29], (const float*)d_in[28], Xe);
  zero_pad_cols<<<NTOK,64,0,stream>>>(Ab);   // ctx pad cols deterministic zero

  // ---------- encoders ----------
  {
    EncParams P;
    P.wenc = WENC;
    P.b1 = (const float*)d_in[36]; P.b2 = (const float*)d_in[37];
    P.bi = (const float*)d_in[38]; P.bk = (const float*)d_in[39];
    P.bo = (const float*)d_in[40]; P.bp = (const float*)d_in[41];
    P.bq = (const float*)d_in[42]; P.bv = (const float*)d_in[43];
    P.g1 = (const float*)d_in[44]; P.g2 = (const float*)d_in[45];
    run_encoder(Xd, DrugMask, P, Qb, Kb, Vb, Ab, Yf, Ib, stream);
  }
  {
    EncParams P;
    P.wenc = WENC + (size_t)NLAYER * LAYER_HALVES;
    P.b1 = (const float*)d_in[52]; P.b2 = (const float*)d_in[53];
    P.bi = (const float*)d_in[54]; P.bk = (const float*)d_in[55];
    P.bo = (const float*)d_in[56]; P.bp = (const float*)d_in[57];
    P.bq = (const float*)d_in[58]; P.bv = (const float*)d_in[59];
    P.g1 = (const float*)d_in[60]; P.g2 = (const float*)d_in[61];
    run_encoder(Xe, SEMask, P, Qb, Kb, Vb, Ab, Yf, Ib, stream);
  }

  // ---------- interaction + conv + MLP head ----------
  interaction_kernel<<<BATCH,256,0,stream>>>(Xd, Xe, iv);
  conv3_kernel<<<BATCH,256,0,stream>>>(iv, conv_w, conv_b, cvo);

  gemm_bf16<0,1><<<dim3(512/GBN, BATCH/GBM),128,0,stream>>>(cvo, CONVN, W1bf, 512,
                                                            h1, 512, b1f, 512,
                                                            BATCH, 512, CONVN);
  bn_kernel<1><<<(BATCH*512)/256,256,0,stream>>>(h1, bn1_g, bn1_b, bn1_m, bn1_v,
                                                 h1bn, 512, BATCH*512);
  gemm_bf16<0,1><<<dim3(1, BATCH/GBM),128,0,stream>>>(h1bn, 512, W2bf, 64,
                                                      h2, 64, b2f, 64,
                                                      BATCH, 64, 512);
  bn_kernel<0><<<(BATCH*64)/256,256,0,stream>>>(h2, bn2_g, bn2_b, bn2_m, bn2_v,
                                                h2bn, 64, BATCH*64);
  head_tail<<<BATCH,64,0,stream>>>(h2bn, W3, b3f, W4, b4f, (float*)d_out);
}
